// Net_23931557773461
// MI455X (gfx1250) — matrix operand order repaired and checked by reading, not compiled, not run on hardware
//
#include <hip/hip_runtime.h>
#include <hip/hip_bf16.h>

typedef __attribute__((ext_vector_type(16))) __bf16 v16bf;
typedef __attribute__((ext_vector_type(8)))  float  v8f;
typedef int v4i __attribute__((vector_size(16)));
typedef __attribute__((address_space(1))) v4i as1_v4i;  // global (HIP __device__)
typedef __attribute__((address_space(3))) v4i as3_v4i;  // LDS

#if __has_builtin(__builtin_amdgcn_global_load_async_to_lds_b128)
#define HAS_ASYNC_LDS 1
#else
#define HAS_ASYNC_LDS 0
#endif

__device__ __forceinline__ void wait_asynccnt0() {
#if __has_builtin(__builtin_amdgcn_s_wait_asynccnt)
  __builtin_amdgcn_s_wait_asynccnt(0);
#else
  asm volatile("s_wait_asynccnt 0x0" ::: "memory");
#endif
}

// ---------------------------------------------------------------- utilities
__global__ void fill_zero_f(float* __restrict__ p, long n) {
  long i = (long)blockIdx.x * blockDim.x + threadIdx.x;
  if (i < n) p[i] = 0.f;
}

__global__ void degree_kernel(const int* __restrict__ src, const int* __restrict__ dst,
                              float* __restrict__ outdeg, float* __restrict__ indeg, int E) {
  int e = blockIdx.x * blockDim.x + threadIdx.x;
  if (e < E) {
    atomicAdd(&outdeg[src[e]], 1.f);   // exact for counts << 2^24
    atomicAdd(&indeg[dst[e]], 1.f);
  }
}

__global__ void invsqrt_kernel(float* __restrict__ x, int n) {
  int i = blockIdx.x * blockDim.x + threadIdx.x;
  if (i < n) x[i] = rsqrtf(fmaxf(x[i], 1.f));
}

// coef_e = ew_e * rsqrt(outdeg[src]) * rsqrt(indeg[dst])  (folds both DGL norms)
__global__ void coef_kernel(const float* __restrict__ ew, const int* __restrict__ src,
                            const int* __restrict__ dst, const float* __restrict__ iso,
                            const float* __restrict__ isi, float* __restrict__ coef, int E) {
  int e = blockIdx.x * blockDim.x + threadIdx.x;
  if (e < E) coef[e] = ew[e] * iso[src[e]] * isi[dst[e]];
}

// ---------------------------------------------------------------- CSR build (by dst)
__global__ void scan1(const float* __restrict__ deg, int* __restrict__ rowptr,
                      int* __restrict__ bsum, int n) {
  __shared__ int s[256];
  int i = blockIdx.x * 256 + threadIdx.x;
  int v = (i < n) ? (int)deg[i] : 0;
  s[threadIdx.x] = v;
  __syncthreads();
  for (int off = 1; off < 256; off <<= 1) {
    int t = (threadIdx.x >= off) ? s[threadIdx.x - off] : 0;
    __syncthreads();
    s[threadIdx.x] += t;
    __syncthreads();
  }
  if (i < n) rowptr[i] = s[threadIdx.x] - v;  // exclusive, block-local
  if (threadIdx.x == 255) bsum[blockIdx.x] = s[255];
}

__global__ void scan2(int* __restrict__ bsum, int nb, int* __restrict__ rowptr, int n) {
  if (blockIdx.x == 0 && threadIdx.x == 0) {
    int run = 0;
    for (int b = 0; b < nb; ++b) { int t = bsum[b]; bsum[b] = run; run += t; }
    rowptr[n] = run;  // == E
  }
}

__global__ void scan3(int* __restrict__ rowptr, const int* __restrict__ bsum, int n) {
  int i = blockIdx.x * blockDim.x + threadIdx.x;
  if (i < n) rowptr[i] += bsum[i >> 8];
}

__global__ void copy_int(const int* __restrict__ a, int* __restrict__ b, int n) {
  int i = blockIdx.x * blockDim.x + threadIdx.x;
  if (i < n) b[i] = a[i];
}

__global__ void csr_fill(const int* __restrict__ src, const int* __restrict__ dst,
                         const float* __restrict__ coef, int* __restrict__ cursor,
                         int* __restrict__ esrc, float* __restrict__ ecoef, int E) {
  int e = blockIdx.x * blockDim.x + threadIdx.x;
  if (e < E) {
    int p = atomicAdd(&cursor[dst[e]], 1);
    esrc[p] = src[e];
    ecoef[p] = coef[e];
  }
}

// ---------------------------------------------------------------- aggregation (no atomics)
// F=128, h stored bf16: one wave32 per node, 4 feats per lane in registers.
__global__ void __launch_bounds__(256)
agg128(const __bf16* __restrict__ h, const int* __restrict__ rowptr,
       const int* __restrict__ esrc, const float* __restrict__ ecoef,
       float* __restrict__ agg, int N) {
  int wave = threadIdx.x >> 5, lane = threadIdx.x & 31;
  int node = blockIdx.x * 8 + wave;
  if (node >= N) return;
  int beg = rowptr[node], end = rowptr[node + 1];
  const int f0 = lane * 4;
  float a0 = 0.f, a1 = 0.f, a2 = 0.f, a3 = 0.f;
  for (int j = beg; j < end; ++j) {
    if (j + 4 < end)  // pull upcoming source rows toward L2/WGP$ (global_prefetch_b8)
      __builtin_prefetch(h + (long)esrc[j + 4] * 128 + f0, 0, 3);
    float c = ecoef[j];
    const __bf16* hp = h + (long)esrc[j] * 128 + f0;   // 8B b64 gather per lane
    a0 += c * (float)hp[0];
    a1 += c * (float)hp[1];
    a2 += c * (float)hp[2];
    a3 += c * (float)hp[3];
  }
  float4 r = {a0, a1, a2, a3};
  *(float4*)(agg + (long)node * 128 + f0) = r;
}

// F=16 (first layer), node_feat f32: one half-wave per node.
__global__ void agg16(const float* __restrict__ x, const int* __restrict__ rowptr,
                      const int* __restrict__ esrc, const float* __restrict__ ecoef,
                      float* __restrict__ agg, int N) {
  long tid = (long)blockIdx.x * blockDim.x + threadIdx.x;
  int node = (int)(tid >> 4);
  int f = (int)(tid & 15);
  if (node >= N) return;
  int beg = rowptr[node], end = rowptr[node + 1];
  float a = 0.f;
  for (int j = beg; j < end; ++j) a += ecoef[j] * x[(long)esrc[j] * 16 + f];
  agg[(long)node * 16 + f] = a;
}

// ---------------------------------------------------------------- weight packing
// Pack W [Kin x Nout] (row-major) into B fragments for v_wmma_f32_16x16x32_bf16:
// frag (ks, tile): lane<16 -> col = tile*16+lane,    elem i <-> K = ks*32 + i
//                  lane>=16 -> col = tile*16+lane-16, elem i <-> K = ks*32 + 16 + i
__global__ void pack_w(const float* __restrict__ W, __bf16* __restrict__ out,
                       int Kin, int Nout, int ksteps, int ntiles) {
  int frags = ksteps * ntiles * 32 * 16;
  int i = blockIdx.x * blockDim.x + threadIdx.x;
  if (i >= frags) return;
  int layer = blockIdx.y;
  const float* Wl = W + (long)layer * Kin * Nout;
  __bf16* ol = out + (long)layer * frags;
  int elem = i & 15;
  int lane = (i >> 4) & 31;
  int kt = i >> 9;  // ks*ntiles + tile
  int tile = kt % ntiles;
  int ks = kt / ntiles;
  int col = tile * 16 + (lane & 15);
  int k = ks * 32 + ((lane >= 16) ? 16 : 0) + elem;
  float v = (k < Kin && col < Nout) ? Wl[(long)k * Nout + col] : 0.f;
  ol[i] = (__bf16)v;
}

// ---------------------------------------------------------------- WMMA GEMM
// out[Nrows x Nout] = act(A[Nrows x KIN] @ W + bias). One wave -> 16 rows x NTILES*16 cols.
// 4 waves/block; packed W staged once into LDS (async-to-LDS when available).
// Nrows divisible by 16 -> active waves are full tiles (EXEC all ones through WMMA).
template <int KSTEPS, int NTILES, int KIN, bool A_BF16, bool OUT_BF16>
__global__ void __launch_bounds__(128)
gemm_bias_act(const void* __restrict__ Ap, const __bf16* __restrict__ Wp,
              const float* __restrict__ bias, void* __restrict__ outp,
              int Nrows, int Nout, int leaky) {
  __shared__ __align__(32) __bf16 sW[KSTEPS * NTILES * 512];
#if HAS_ASYNC_LDS
  {  // ASYNCcnt-tracked DMA of packed W into LDS, no VGPR round-trip
    for (int c = threadIdx.x; c < KSTEPS * NTILES * 64; c += 128) {
      as1_v4i* g = (as1_v4i*)((const v4i*)Wp + c);
      as3_v4i* l = (as3_v4i*)((v4i*)sW + c);
      __builtin_amdgcn_global_load_async_to_lds_b128(g, l, 0, 0);
    }
    wait_asynccnt0();
  }
#else
  {  // fallback: cooperative register copy
    float4* d = (float4*)sW;
    const float4* s = (const float4*)Wp;
    for (int c = threadIdx.x; c < KSTEPS * NTILES * 64; c += 128) d[c] = s[c];
  }
#endif
  __syncthreads();

  const int lane = threadIdx.x & 31;
  const int wave = threadIdx.x >> 5;
  const int rowBase = (blockIdx.x * 4 + wave) * 16;
  if (rowBase >= Nrows) return;  // wave-uniform exit (after barrier)
  const int laneLo = lane & 15;
  const int hi8 = (lane >> 4) << 3;  // 0 or 8
  const long arow = (long)(rowBase + laneLo) * KIN;

  v8f acc[NTILES] = {};

#pragma unroll
  for (int ks = 0; ks < KSTEPS; ++ks) {
    const int k0 = ks * 32;
    // A fragment: contiguous K layout (lane half h: elem i <-> K+16h+i), matching packed B
    v16bf a;
#pragma unroll
    for (int i = 0; i < 16; ++i) {
      int k = k0 + 2 * hi8 + i;
      float f = 0.f;
      if (KSTEPS * 32 <= KIN || k < KIN) {
        f = A_BF16 ? (float)((const __bf16*)Ap)[arow + k]
                   : ((const float*)Ap)[arow + k];
      }
      a[i] = (__bf16)f;
    }
    // Batch all B-fragment ds_loads for this k-step, then issue the WMMA chain,
    // so waits graduate instead of full-stalling before every WMMA.
    v16bf bf[NTILES];
#pragma unroll
    for (int t = 0; t < NTILES; ++t)
      bf[t] = *(const v16bf*)(sW + ((ks * NTILES + t) * 32 + lane) * 16);
#pragma unroll
    for (int t = 0; t < NTILES; ++t)
      acc[t] = __builtin_amdgcn_wmma_f32_16x16x32_bf16(
          /*neg_a=*/false, a, /*neg_b=*/false, bf[t],
          /*c_mod=*/(short)0, acc[t], /*reuse_a=*/false, /*reuse_b=*/false);
  }

  // C/D layout: VGPR r, lanes 0-15 -> (M=r, N=lane); lanes 16-31 -> (M=8+r, N=lane-16)
  const int rOff = (lane >> 4) * 8;
#pragma unroll
  for (int t = 0; t < NTILES; ++t) {
    const int col = t * 16 + laneLo;
    if (col < Nout) {
      const float bb = bias[col];
#pragma unroll
      for (int r = 0; r < 8; ++r) {
        float v = acc[t][r] + bb;
        if (leaky) v = (v >= 0.f) ? v : 0.01f * v;
        const long idx = (long)(rowBase + rOff + r) * Nout + col;
        if (OUT_BF16) ((__bf16*)outp)[idx] = (__bf16)v;
        else          ((float*)outp)[idx] = v;
      }
    }
  }
}

// ---------------------------------------------------------------- launch
static inline size_t alignUp(size_t x) { return (x + 255) & ~(size_t)255; }

extern "C" void kernel_launch(void* const* d_in, const int* in_sizes, int n_in,
                              void* d_out, int out_size, void* d_ws, size_t ws_size,
                              hipStream_t stream) {
  const float* node_feat = (const float*)d_in[0];  // [N,16]
  const float* ew        = (const float*)d_in[1];  // [E]
  const int*   src       = (const int*)  d_in[2];  // [E]
  const int*   dst       = (const int*)  d_in[3];  // [E]
  const float* W1        = (const float*)d_in[4];  // [16,128]
  const float* b1        = (const float*)d_in[5];  // [128]
  const float* W_stack   = (const float*)d_in[6];  // [L,128,128]
  const float* b_stack   = (const float*)d_in[7];  // [L,128]
  const float* fc_W      = (const float*)d_in[8];  // [128,4]
  const float* fc_b      = (const float*)d_in[9];  // [4]
  float* out = (float*)d_out;

  const int N = in_sizes[0] / 16;
  const int E = in_sizes[1];
  const int L = in_sizes[6] / (128 * 128);  // 19
  const int NB = (N + 255) / 256;           // scan blocks

  // workspace carve-up
  char* p = (char*)d_ws;
  float* iso    = (float*)p; p += alignUp((size_t)N * 4);
  float* isi    = (float*)p; p += alignUp((size_t)N * 4);
  float* coef   = (float*)p; p += alignUp((size_t)E * 4);
  int*   rowptr = (int*)p;   p += alignUp((size_t)(N + 1) * 4);
  int*   bsum   = (int*)p;   p += alignUp((size_t)NB * 4);
  int*   cursor = (int*)p;   p += alignUp((size_t)N * 4);
  int*   esrc   = (int*)p;   p += alignUp((size_t)E * 4);
  float* ecoef  = (float*)p; p += alignUp((size_t)E * 4);
  float* agg    = (float*)p; p += alignUp((size_t)N * 128 * 4);   // f32 GEMM input
  __bf16* h     = (__bf16*)p; p += alignUp((size_t)N * 128 * 2);  // bf16 activations
  const long packHidden = 4L * 8 * 32 * 16;  // bf16 elems / hidden layer
  __bf16* pW1 = (__bf16*)p; p += alignUp(1L * 8 * 32 * 16 * 2);
  __bf16* pWs = (__bf16*)p; p += alignUp((size_t)L * packHidden * 2);
  __bf16* pFc = (__bf16*)p; p += alignUp(4L * 1 * 32 * 16 * 2);

  const int T = 256;
  auto blocks = [](long n, int t) { return (int)((n + t - 1) / t); };

  // degrees (raw) -> CSR rowptr -> norms -> per-edge coefficient -> CSR fill
  fill_zero_f<<<blocks(N, T), T, 0, stream>>>(iso, N);
  fill_zero_f<<<blocks(N, T), T, 0, stream>>>(isi, N);
  degree_kernel<<<blocks(E, T), T, 0, stream>>>(src, dst, iso, isi, E);
  scan1<<<NB, 256, 0, stream>>>(isi, rowptr, bsum, N);   // isi still holds raw in-degree
  scan2<<<1, 32, 0, stream>>>(bsum, NB, rowptr, N);
  scan3<<<blocks(N, T), T, 0, stream>>>(rowptr, bsum, N);
  invsqrt_kernel<<<blocks(N, T), T, 0, stream>>>(iso, N);
  invsqrt_kernel<<<blocks(N, T), T, 0, stream>>>(isi, N);
  coef_kernel<<<blocks(E, T), T, 0, stream>>>(ew, src, dst, iso, isi, coef, E);
  copy_int<<<blocks(N, T), T, 0, stream>>>(rowptr, cursor, N);
  csr_fill<<<blocks(E, T), T, 0, stream>>>(src, dst, coef, cursor, esrc, ecoef, E);

  // pack weights to bf16 WMMA B-fragments
  pack_w<<<dim3(blocks(1L * 8 * 512, T), 1), T, 0, stream>>>(W1, pW1, 16, 128, 1, 8);
  pack_w<<<dim3(blocks(packHidden, T), L), T, 0, stream>>>(W_stack, pWs, 128, 128, 4, 8);
  pack_w<<<dim3(blocks(4L * 1 * 512, T), 1), T, 0, stream>>>(fc_W, pFc, 128, 4, 4, 1);

  const int gemmBlocks = blocks((long)N, 64);  // 4 waves x 16 rows per block

  // layer 1: [N,16] conv -> h[N,128] (bf16)
  agg16<<<blocks((long)N * 16, T), T, 0, stream>>>(node_feat, rowptr, esrc, ecoef, agg, N);
  gemm_bias_act<1, 8, 16, false, true><<<gemmBlocks, 128, 0, stream>>>(
      agg, pW1, b1, h, N, 128, 1);

  // hidden layers 2..20: atomic-free CSR aggregation + WMMA GEMM
  for (int l = 0; l < L; ++l) {
    agg128<<<blocks((long)N, 8), 256, 0, stream>>>(h, rowptr, esrc, ecoef, agg, N);
    gemm_bias_act<4, 8, 128, false, true><<<gemmBlocks, 128, 0, stream>>>(
        agg, pWs + (long)l * packHidden, b_stack + (long)l * 128, h, N, 128, 1);
  }

  // final FC: h[N,128] (bf16) -> out[N,4] (f32), no activation
  gemm_bias_act<4, 1, 128, true, false><<<gemmBlocks, 128, 0, stream>>>(
      h, pFc, fc_b, out, N, 4, 0);
}
